// SelfAttention_858993459747
// MI455X (gfx1250) — matrix operand order repaired and checked by reading, not compiled, not run on hardware
//
#include <hip/hip_runtime.h>
#include <hip/hip_bf16.h>
#include <math.h>

// ---------------------------------------------------------------------------
// MHA forward for MI455X (gfx1250): wave32, V_WMMA_F32_16X16X32_BF16 for all
// GEMMs, TDM (tensor_load_to_lds + s_wait_tensorcnt) double-buffered K/V
// staging in the attention kernel.
// N=2, L=2048, E=1024, H=16, D=64.
// ---------------------------------------------------------------------------

#define NN   2
#define LL   2048
#define EE   1024
#define HH   16
#define DD   64

typedef __bf16 v16bf __attribute__((ext_vector_type(16)));
typedef float  v8f   __attribute__((ext_vector_type(8)));
typedef unsigned int v4u __attribute__((ext_vector_type(4)));
typedef int v8i __attribute__((ext_vector_type(8)));
typedef int v4i __attribute__((ext_vector_type(4)));

union BF16Frag {
    v16bf v;
    uint4 q[2];
    unsigned short us[16];
};

__device__ __forceinline__ unsigned short f32_to_bf16(float f) {
    union { float f; unsigned int u; } a; a.f = f;
    unsigned int u = a.u;
    unsigned int r = (u + 0x7FFFu + ((u >> 16) & 1u)) >> 16;   // RNE
    return (unsigned short)r;
}

__device__ __forceinline__ v8f wmma_bf16(v16bf a, v16bf b, v8f c) {
    return __builtin_amdgcn_wmma_f32_16x16x32_bf16(
        false, a, false, b, (short)0, c, false, false);
}

// A fragment (16x32 bf16): row = lane&15; halves i -> K = koff+i,
// koff = (lane<16)?0:16.
__device__ __forceinline__ v16bf load_a_bf16(const unsigned short* base, int lane) {
    int koff = (lane < 16) ? 0 : 16;
    BF16Frag f;
    f.q[0] = *(const uint4*)(base + koff);
    f.q[1] = *(const uint4*)(base + koff + 8);
    return f.v;
}

__device__ __forceinline__ v16bf load_a_f32(const float* base, int lane) {
    int koff = (lane < 16) ? 0 : 16;
    BF16Frag f;
#pragma unroll
    for (int i = 0; i < 8; ++i) f.us[i]     = f32_to_bf16(base[koff + i]);
#pragma unroll
    for (int i = 0; i < 8; ++i) f.us[8 + i] = f32_to_bf16(base[koff + 8 + i]);
    return f.v;
}

// B fragment (32x16 bf16): lane holds col N=lane&15; halves i -> K =
// (lane<16?0:16)+i.  `base` already includes the +0/+16 lane split.
__device__ __forceinline__ v16bf load_b_bf16(const unsigned short* base) {
    BF16Frag f;
    f.q[0] = *(const uint4*)(base);
    f.q[1] = *(const uint4*)(base + 8);
    return f.v;
}

// Wave-relative LDS byte address: per ISA §10.2 the LDS aperture mapping is
// the low 32 address bits, so the low 32 bits of a generic shared pointer are
// the logical LDS offset the TDM descriptor wants.
__device__ __forceinline__ unsigned lds_addr_of(const void* p) {
    return (unsigned)(unsigned long long)p;
}

#if defined(__has_builtin)
#if __has_builtin(__builtin_amdgcn_tensor_load_to_lds)
#define HAVE_TDM 1
#endif
#endif

#ifdef HAVE_TDM
// Build D# per ISA §8.3/8.4 and issue TENSOR_LOAD_TO_LDS (2D tile, 2-byte
// elements).  tile1==0 -> 1D copy of tile0 elements.  This toolchain exposes
// the 6-arg builtin: (g0 v4u, g1 v8i, g2 v4i, g3 v4i, extra v8i, cpol i32);
// groups 2/3 + extra are only needed for >2D tensors -> zero-filled.
__device__ __forceinline__ void tdm_load_2d(unsigned lds_off, const void* gptr,
                                            unsigned tile0, unsigned tile1,
                                            unsigned dim0, unsigned dim1,
                                            unsigned stride0) {
    unsigned long long ga = (unsigned long long)gptr;
    v4u g0;
    g0[0] = 1u;                                   // count=1, user mode
    g0[1] = lds_off;                              // lds_addr (bytes)
    g0[2] = (unsigned)ga;                         // global_addr[31:0]
    g0[3] = ((unsigned)(ga >> 32) & 0x01FFFFFFu)  // global_addr[56:32]
            | 0x80000000u;                        // type=2 ("image") @ [127:126]
    v8i g1;
    g1[0] = (int)(1u << 16);                      // data_size=1 -> 2 bytes
    g1[1] = (int)((dim0 & 0xFFFFu) << 16);        // tensor_dim0[15:0] @ [63:48]
    g1[2] = (int)(((dim0 >> 16) & 0xFFFFu)        // tensor_dim0[31:16]
            | ((dim1 & 0xFFFFu) << 16));          // tensor_dim1[15:0]
    g1[3] = (int)(((dim1 >> 16) & 0xFFFFu)        // tensor_dim1[31:16]
            | ((tile0 & 0xFFFFu) << 16));         // tile_dim0 @ [127:112]
    g1[4] = (int)(tile1 & 0xFFFFu);               // tile_dim1, tile_dim2=0
    g1[5] = (int)stride0;                         // tensor_dim0_stride[31:0]
    g1[6] = 0;                                    // stride0[47:32], stride1 lo
    g1[7] = 0;                                    // stride1 hi
    v4i z4 = {};
    v8i z8 = {};
    __builtin_amdgcn_tensor_load_to_lds(g0, g1, z4, z4, z8, 0);
}
#endif

// ---------------------------------------------------------------------------
// Kernel 1: weights f32 -> bf16.
// ---------------------------------------------------------------------------
__global__ void k_cvt_weights(const float* __restrict__ Wv,
                              const float* __restrict__ Wk,
                              const float* __restrict__ Wq,
                              const float* __restrict__ Wo,
                              unsigned short* __restrict__ Whb,
                              unsigned short* __restrict__ Wob) {
    int i = blockIdx.x * 256 + threadIdx.x;
    if (i < DD * DD) {
        Whb[i]               = f32_to_bf16(Wv[i]);
        Whb[DD * DD + i]     = f32_to_bf16(Wk[i]);
        Whb[2 * DD * DD + i] = f32_to_bf16(Wq[i]);
    }
    if (i < EE * EE) Wob[i] = f32_to_bf16(Wo[i]);
}

// ---------------------------------------------------------------------------
// Kernel 2: per-head projections (x @ W^T).  Q,K -> [n,h,l,d]; V -> [n,h,d,l].
// ---------------------------------------------------------------------------
__global__ void k_qkv_proj(const float* __restrict__ vals,
                           const float* __restrict__ keys,
                           const float* __restrict__ query,
                           const unsigned short* __restrict__ Whb,
                           unsigned short* __restrict__ Qp,
                           unsigned short* __restrict__ Kp,
                           unsigned short* __restrict__ Vt) {
    int w    = threadIdx.x >> 5;
    int lane = threadIdx.x & 31;
    int unit = blockIdx.x * 8 + w;          // 3*2*16*128 = 12288 units
    int lt = unit & 127;
    int h  = (unit >> 7) & 15;
    int n  = (unit >> 11) & 1;
    int t  = unit >> 12;                    // 0=v, 1=k, 2=q
    int lanelo = lane & 15, hi = lane >> 4;

    const float* X = (t == 0) ? vals : (t == 1 ? keys : query);
    const unsigned short* W = Whb + t * DD * DD;

    int l = lt * 16 + lanelo;
    const float* xrow = X + ((size_t)(n * LL + l)) * EE + h * DD;
    v16bf A0 = load_a_f32(xrow, lane);
    v16bf A1 = load_a_f32(xrow + 32, lane);

    v8f C[4];
#pragma unroll
    for (int et = 0; et < 4; ++et) {
        int e = et * 16 + lanelo;
        v16bf B0 = load_b_bf16(W + (size_t)e * DD + hi * 16);
        v16bf B1 = load_b_bf16(W + (size_t)e * DD + 32 + hi * 16);
        v8f c = {};
        c = wmma_bf16(A0, B0, c);
        c = wmma_bf16(A1, B1, c);
        C[et] = c;
    }

    int nh = n * HH + h;
#pragma unroll
    for (int et = 0; et < 4; ++et) {
        int e = et * 16 + lanelo;
#pragma unroll
        for (int r = 0; r < 8; ++r) {
            int lr = lt * 16 + r + hi * 8;
            unsigned short b = f32_to_bf16(C[et][r]);
            if (t == 2)      Qp[((size_t)nh * LL + lr) * DD + e] = b;
            else if (t == 1) Kp[((size_t)nh * LL + lr) * DD + e] = b;
            else             Vt[((size_t)nh * DD + e) * LL + lr] = b;
        }
    }
}

// ---------------------------------------------------------------------------
// Kernel 3: flash attention.  Block = 8 waves, all on the SAME (n,h) and 8
// consecutive 16-row query tiles, so K/V 32-key tiles are shared 8x.  TDM
// stages K (1D, 4KB) and V (2D: 64 rows x 32 cols of a 2048-wide tensor)
// into double-buffered LDS; wave 0 issues, TENSORcnt + barrier synchronize.
// ---------------------------------------------------------------------------
__global__ __launch_bounds__(256) void k_attention(
        const unsigned short* __restrict__ Qp,
        const unsigned short* __restrict__ Kp,
        const unsigned short* __restrict__ Vt,
        const int* __restrict__ mask,
        unsigned short* __restrict__ attnb) {
    __shared__ __align__(16) unsigned short kbuf[2][32 * DD];   // [key][d]
    __shared__ __align__(16) unsigned short vbuf[2][DD * 32];   // [d][key]
    __shared__ __align__(16) unsigned short pstage[8][16][48];

    int w    = threadIdx.x >> 5;
    int lane = threadIdx.x & 31;
    int unit = blockIdx.x * 8 + w;          // 2*16*128 = 4096 units
    int qt = unit & 127;
    int h  = (unit >> 7) & 15;
    int n  = unit >> 11;
    int nh = n * HH + h;                    // identical for all 8 waves
    int lanelo = lane & 15, hi = lane >> 4;

    const unsigned short* Kbase = Kp + (size_t)nh * LL * DD;
    const unsigned short* Vbase = Vt + (size_t)nh * DD * LL;

    // Q tile (16x64), resident A fragments.
    const unsigned short* qrow = Qp + ((size_t)nh * LL + qt * 16 + lanelo) * DD;
    v16bf Qa0 = load_a_bf16(qrow, lane);
    v16bf Qa1 = load_a_bf16(qrow + 32, lane);

    float m[8], lsum[8];
    v8f O[4];
#pragma unroll
    for (int r = 0; r < 8; ++r) { m[r] = -INFINITY; lsum[r] = 0.0f; }
#pragma unroll
    for (int dt = 0; dt < 4; ++dt) { v8f z = {}; O[dt] = z; }

    const float inv_sqrt_e = 0.03125f;      // 1/sqrt(1024)
    const int NT = LL / 32;                 // 64 key tiles

#ifdef HAVE_TDM
    if (w == 0) {   // prologue: stage tile 0 into buffer 0
        tdm_load_2d(lds_addr_of(&kbuf[0][0]), Kbase, 32 * DD, 0,
                    32 * DD, 1, 32 * DD);                       // 1D 4KB
        tdm_load_2d(lds_addr_of(&vbuf[0][0]), Vbase, 32, DD,
                    LL, DD, LL);                                // 2D tile
    }
#endif

    for (int kb = 0; kb < NT; ++kb) {
        int cur = kb & 1;
        int keyBase = kb * 32;

#ifdef HAVE_TDM
        if (w == 0) {
            if (kb + 1 < NT) {
                int nb = keyBase + 32;
                tdm_load_2d(lds_addr_of(&kbuf[cur ^ 1][0]),
                            Kbase + (size_t)nb * DD, 32 * DD, 0,
                            32 * DD, 1, 32 * DD);
                tdm_load_2d(lds_addr_of(&vbuf[cur ^ 1][0]),
                            Vbase + nb, 32, DD, LL, DD, LL);
                __builtin_amdgcn_s_wait_tensorcnt(2);   // tile kb landed
            } else {
                __builtin_amdgcn_s_wait_tensorcnt(0);
            }
        }
        __syncthreads();
#else
        {   // fallback: cooperative synchronous staging
            int tid = threadIdx.x;
            const uint4* gk = (const uint4*)(Kbase + (size_t)keyBase * DD);
            const uint4* gv0 = (const uint4*)(Vbase + keyBase);
            uint4* lk = (uint4*)&kbuf[cur][0];
            lk[tid] = gk[tid];                          // 256 * 16B = 4KB
            int row = tid >> 2, c4 = tid & 3;           // 64 rows x 4 uint4
            ((uint4*)&vbuf[cur][0])[tid] =
                *(const uint4*)((const unsigned short*)gv0 + (size_t)row * LL + c4 * 8);
            __syncthreads();
        }
#endif

        // ---- S = Q K^T (two 16-key subtiles), fragments from LDS -------
        v8f S[2];
#pragma unroll
        for (int kt = 0; kt < 2; ++kt) {
            const unsigned short* krow =
                &kbuf[cur][(kt * 16 + lanelo) * DD + hi * 16];
            v16bf Kb0 = load_b_bf16(krow);
            v16bf Kb1 = load_b_bf16(krow + 32);
            v8f s = {};
            s = wmma_bf16(Qa0, Kb0, s);
            s = wmma_bf16(Qa1, Kb1, s);
            int mv = mask[n * LL + keyBase + kt * 16 + lanelo];
#pragma unroll
            for (int r = 0; r < 8; ++r) {
                float x = s[r];
                x = (mv == 0) ? -1e20f : x;             // mask BEFORE scale
                s[r] = x * inv_sqrt_e;
            }
            S[kt] = s;
        }

        // ---- online softmax (rows live in 16-lane groups) --------------
        float c[8];
#pragma unroll
        for (int r = 0; r < 8; ++r) {
            float v = fmaxf(S[0][r], S[1][r]);
            v = fmaxf(v, __shfl_xor(v, 1, 16));
            v = fmaxf(v, __shfl_xor(v, 2, 16));
            v = fmaxf(v, __shfl_xor(v, 4, 16));
            v = fmaxf(v, __shfl_xor(v, 8, 16));
            float mnew = fmaxf(m[r], v);
            c[r] = __expf(m[r] - mnew);
            m[r] = mnew;
            float p0 = __expf(S[0][r] - mnew);
            float p1 = __expf(S[1][r] - mnew);
            S[0][r] = p0; S[1][r] = p1;
            float rs = p0 + p1;
            rs += __shfl_xor(rs, 1, 16);
            rs += __shfl_xor(rs, 2, 16);
            rs += __shfl_xor(rs, 4, 16);
            rs += __shfl_xor(rs, 8, 16);
            lsum[r] = lsum[r] * c[r] + rs;
        }
#pragma unroll
        for (int dt = 0; dt < 4; ++dt)
#pragma unroll
            for (int r = 0; r < 8; ++r) O[dt][r] *= c[r];

        // ---- P: C/D layout -> LDS -> A layout (wave-private) -----------
#pragma unroll
        for (int kt = 0; kt < 2; ++kt)
#pragma unroll
            for (int r = 0; r < 8; ++r)
                pstage[w][r + hi * 8][kt * 16 + lanelo] = f32_to_bf16(S[kt][r]);
        asm volatile("s_wait_dscnt 0" ::: "memory");
        v16bf Pa = load_a_bf16(&pstage[w][lanelo][0], lane);

        // ---- O += P(16x32) * V(32x16) per d tile, V from LDS -----------
#pragma unroll
        for (int dt = 0; dt < 4; ++dt) {
            const unsigned short* vrow =
                &vbuf[cur][(dt * 16 + lanelo) * 32 + hi * 16];
            v16bf Vb = load_b_bf16(vrow);
            O[dt] = wmma_bf16(Pa, Vb, O[dt]);
        }
        __syncthreads();    // everyone done with buf[cur] before it refills
    }

    // ---- epilogue: normalize, store bf16 [n, q, h*64+d] ----------------
#pragma unroll
    for (int r = 0; r < 8; ++r) lsum[r] = 1.0f / lsum[r];
#pragma unroll
    for (int dt = 0; dt < 4; ++dt) {
        int d = dt * 16 + lanelo;
#pragma unroll
        for (int r = 0; r < 8; ++r) {
            int q = qt * 16 + r + hi * 8;
            attnb[((size_t)n * LL + q) * EE + h * DD + d] =
                f32_to_bf16(O[dt][r] * lsum[r]);
        }
    }
}

// ---------------------------------------------------------------------------
// Kernel 4: out = attn @ Wo^T + bo.
// ---------------------------------------------------------------------------
__global__ __launch_bounds__(256) void k_out_proj(
        const unsigned short* __restrict__ attnb,
        const unsigned short* __restrict__ Wob,
        const float* __restrict__ bo,
        float* __restrict__ out) {
    int w    = threadIdx.x >> 5;
    int lane = threadIdx.x & 31;
    int unit = blockIdx.x * 8 + w;
    int et4 = unit & 15;
    int rt  = unit >> 4;
    int lanelo = lane & 15, hi = lane >> 4;

    const unsigned short* arow = attnb + (size_t)(rt * 16 + lanelo) * EE;

    v8f C[4];
#pragma unroll
    for (int et = 0; et < 4; ++et) { v8f z = {}; C[et] = z; }

    for (int ks = 0; ks < EE / 32; ++ks) {
        v16bf A = load_a_bf16(arow + ks * 32, lane);
#pragma unroll
        for (int et = 0; et < 4; ++et) {
            int e = et4 * 64 + et * 16 + lanelo;
            const unsigned short* wrow = Wob + (size_t)e * EE + ks * 32 + hi * 16;
            v16bf B = load_b_bf16(wrow);
            C[et] = wmma_bf16(A, B, C[et]);
        }
    }

#pragma unroll
    for (int et = 0; et < 4; ++et) {
        int e = et4 * 64 + et * 16 + lanelo;
        float be = bo[e];
#pragma unroll
        for (int r = 0; r < 8; ++r) {
            int row = rt * 16 + r + hi * 8;
            out[(size_t)row * EE + e] = C[et][r] + be;
        }
    }
}

// ---------------------------------------------------------------------------
// Launch.  Workspace: Qp@0(8M) Kp@8M(8M) Vt@16M(8M) attn@24M(8M)
//                     Wob@32M(2M) Whb@34M(24K)
// ---------------------------------------------------------------------------
extern "C" void kernel_launch(void* const* d_in, const int* in_sizes, int n_in,
                              void* d_out, int out_size, void* d_ws, size_t ws_size,
                              hipStream_t stream) {
    (void)in_sizes; (void)n_in; (void)out_size; (void)ws_size;

    const float* vals  = (const float*)d_in[0];
    const float* keys  = (const float*)d_in[1];
    const float* query = (const float*)d_in[2];
    const int*   mask  = (const int*)d_in[3];
    const float* Wv    = (const float*)d_in[4];
    const float* Wk    = (const float*)d_in[5];
    const float* Wq    = (const float*)d_in[6];
    const float* Wo    = (const float*)d_in[7];
    const float* bo    = (const float*)d_in[8];
    float* out = (float*)d_out;

    char* ws = (char*)d_ws;
    const size_t MB = 1024 * 1024;
    unsigned short* Qp   = (unsigned short*)(ws + 0);
    unsigned short* Kp   = (unsigned short*)(ws + 8 * MB);
    unsigned short* Vt   = (unsigned short*)(ws + 16 * MB);
    unsigned short* attn = (unsigned short*)(ws + 24 * MB);
    unsigned short* Wob  = (unsigned short*)(ws + 32 * MB);
    unsigned short* Whb  = (unsigned short*)(ws + 34 * MB);

    k_cvt_weights<<<dim3((EE * EE + 255) / 256), dim3(256), 0, stream>>>(
        Wv, Wk, Wq, Wo, Whb, Wob);

    k_qkv_proj<<<dim3(12288 / 8), dim3(256), 0, stream>>>(
        vals, keys, query, Whb, Qp, Kp, Vt);

    k_attention<<<dim3(4096 / 8), dim3(256), 0, stream>>>(
        Qp, Kp, Vt, mask, attn);

    k_out_proj<<<dim3(4096 / 8), dim3(256), 0, stream>>>(
        attn, Wob, bo, out);
}
